// GAT_79164837200026
// MI455X (gfx1250) — matrix operand order repaired and checked by reading, not compiled, not run on hardware
//
#include <hip/hip_runtime.h>

// ---------------------------------------------------------------------------
// GAT (2x GATConv, heads=1) for MI455X / gfx1250, wave32.
// GEMMs run on v_wmma_f32_16x16x32_bf16 with split-precision (hi/lo bf16)
// 3-product accumulation to retain near-fp32 accuracy.
// Edge softmax uses ordered-uint atomicMax + f32 atomicAdd (L2-resident).
// ---------------------------------------------------------------------------

typedef __attribute__((ext_vector_type(16))) __bf16 v16bf;
typedef __attribute__((ext_vector_type(8)))  float  v8f;

static constexpr int N_NODES = 100000;
static constexpr int N_EDGES = 1600000;
static constexpr int HF      = 32;            // H_FEATS
static constexpr int NF      = N_NODES * HF;  // node-feature elements

// ------------------------- helpers -----------------------------------------

__device__ __forceinline__ unsigned f2ord(float f) {
    unsigned u = __float_as_uint(f);
    return (u & 0x80000000u) ? ~u : (u | 0x80000000u);
}
__device__ __forceinline__ float ord2f(unsigned k) {
    unsigned v = (k & 0x80000000u) ? (k & 0x7fffffffu) : ~k;
    return __uint_as_float(v);
}

// ------------------------- GEMM: F = X @ W  (W is [KDIM x 32]) --------------
// One wave per 16-row tile; both 16-col output tiles per wave.
// Split precision: C += Ah*Bh + Ah*Bl + Al*Bh  (6 WMMAs per 32-deep K chunk).

template <int KDIM>
__global__ __launch_bounds__(256)
void gat_gemm_wmma(const float* __restrict__ X, const float* __restrict__ W,
                   float* __restrict__ F, int ntiles) {
    const int wid     = threadIdx.x >> 5;
    const int lane    = threadIdx.x & 31;
    const int tile    = blockIdx.x * 8 + wid;
    if (tile >= ntiles) return;
    const int halfsel = lane >> 4;     // 0: lanes 0-15, 1: lanes 16-31
    const int l15     = lane & 15;
    const int row     = tile * 16 + l15;
    const int klo     = halfsel * 8;   // A-fragment K split per ISA layout

    v8f c0 = {};   // output cols 0..15
    v8f c1 = {};   // output cols 16..31

    const float* xrow = X + (size_t)row * KDIM;

    for (int kk = 0; kk < KDIM; kk += 32) {
        // ---- A fragment (16x32 bf16): hi + lo residual ----
        float4 xa = *(const float4*)(xrow + kk + klo);
        float4 xb = *(const float4*)(xrow + kk + klo + 4);
        float4 xc = *(const float4*)(xrow + kk + klo + 16);
        float4 xd = *(const float4*)(xrow + kk + klo + 20);
        float af[16] = { xa.x, xa.y, xa.z, xa.w, xb.x, xb.y, xb.z, xb.w,
                         xc.x, xc.y, xc.z, xc.w, xd.x, xd.y, xd.z, xd.w };
        v16bf a_hi, a_lo;
#pragma unroll
        for (int e = 0; e < 16; ++e) {
            __bf16 h = (__bf16)af[e];
            a_hi[e] = h;
            a_lo[e] = (__bf16)(af[e] - (float)h);
        }

        // ---- B fragments (32x16 bf16) for both column tiles ----
        const int kbase = kk + halfsel * 8;
        v16bf b_hi0, b_lo0, b_hi1, b_lo1;
#pragma unroll
        for (int e = 0; e < 16; ++e) {
            float w0 = W[(size_t)(kbase + e + ((e >> 3) << 3)) * HF + l15];
            float w1 = W[(size_t)(kbase + e + ((e >> 3) << 3)) * HF + 16 + l15];
            __bf16 h0 = (__bf16)w0;
            __bf16 h1 = (__bf16)w1;
            b_hi0[e] = h0; b_lo0[e] = (__bf16)(w0 - (float)h0);
            b_hi1[e] = h1; b_lo1[e] = (__bf16)(w1 - (float)h1);
        }

        c0 = __builtin_amdgcn_wmma_f32_16x16x32_bf16(false, a_hi, false, b_hi0,
                                                     (short)0, c0, false, false);
        c0 = __builtin_amdgcn_wmma_f32_16x16x32_bf16(false, a_hi, false, b_lo0,
                                                     (short)0, c0, false, false);
        c0 = __builtin_amdgcn_wmma_f32_16x16x32_bf16(false, a_lo, false, b_hi0,
                                                     (short)0, c0, false, false);

        c1 = __builtin_amdgcn_wmma_f32_16x16x32_bf16(false, a_hi, false, b_hi1,
                                                     (short)0, c1, false, false);
        c1 = __builtin_amdgcn_wmma_f32_16x16x32_bf16(false, a_hi, false, b_lo1,
                                                     (short)0, c1, false, false);
        c1 = __builtin_amdgcn_wmma_f32_16x16x32_bf16(false, a_lo, false, b_hi1,
                                                     (short)0, c1, false, false);
    }

    // ---- C layout: VGPR v -> row (v + halfsel*8), col = l15 (+16) ----
#pragma unroll
    for (int v = 0; v < 8; ++v) {
        int r = tile * 16 + v + halfsel * 8;
        F[(size_t)r * HF + l15]      = c0[v];
        F[(size_t)r * HF + 16 + l15] = c1[v];
    }
}

// ------------------------- per-node attention dots --------------------------

__global__ void gat_node_dot(const float* __restrict__ F,
                             const float* __restrict__ al,
                             const float* __restrict__ ar,
                             float* __restrict__ el, float* __restrict__ er,
                             int n) {
    int i = blockIdx.x * blockDim.x + threadIdx.x;
    if (i >= n) return;
    const float4* f4 = (const float4*)(F + (size_t)i * HF);
    float sl = 0.f, sr = 0.f;
#pragma unroll
    for (int q = 0; q < 8; ++q) {
        float4 v = f4[q];
        sl += v.x * al[q*4+0] + v.y * al[q*4+1] + v.z * al[q*4+2] + v.w * al[q*4+3];
        sr += v.x * ar[q*4+0] + v.y * ar[q*4+1] + v.z * ar[q*4+2] + v.w * ar[q*4+3];
    }
    el[i] = sl;
    er[i] = sr;
}

// ------------------------- edge score + segment max -------------------------

__global__ void gat_edge_score(const int* __restrict__ src,
                               const int* __restrict__ dst,
                               const float* __restrict__ el,
                               const float* __restrict__ er,
                               float* __restrict__ ebuf,
                               unsigned* __restrict__ mkey, int E) {
    int e = blockIdx.x * blockDim.x + threadIdx.x;
    if (e >= E) return;
    __builtin_prefetch(src + e + 4096, 0, 0);   // global_prefetch_b8
    __builtin_prefetch(dst + e + 4096, 0, 0);
    int s = src[e], d = dst[e];
    float v = el[s] + er[d];
    v = (v > 0.f) ? v : 0.2f * v;               // leaky_relu(0.2)
    ebuf[e] = v;
    atomicMax(mkey + d, f2ord(v));
}

// ------------------------- decode segment max (neutralize empty) ------------

__global__ void gat_decode_m(const unsigned* __restrict__ mkey,
                             float* __restrict__ m, int n) {
    int i = blockIdx.x * blockDim.x + threadIdx.x;
    if (i >= n) return;
    float f = ord2f(mkey[i]);
    m[i] = __builtin_isfinite(f) ? f : 0.f;     // empty segment -> 0
}

// ------------------------- exp + segment sum (denominator) ------------------

__global__ void gat_edge_expsum(const int* __restrict__ dst,
                                float* __restrict__ ebuf,
                                const float* __restrict__ m,
                                float* __restrict__ denom, int E) {
    int e = blockIdx.x * blockDim.x + threadIdx.x;
    if (e >= E) return;
    int d = dst[e];
    float ex = __expf(ebuf[e] - m[d]);
    ebuf[e] = ex;
    atomicAdd(denom + d, ex);
}

// ------------------------- weighted scatter: out[dst] += alpha*feat[src] ----
// 8 lanes per edge; each lane gathers a float4 and issues 4 f32 atomics.

__global__ void gat_edge_scatter(const int* __restrict__ src,
                                 const int* __restrict__ dst,
                                 const float* __restrict__ ebuf,
                                 const float* __restrict__ denom,
                                 const float* __restrict__ F,
                                 float* __restrict__ acc, int E) {
    int t = blockIdx.x * blockDim.x + threadIdx.x;
    if (t >= E * 8) return;
    int e = t >> 3;
    int j = t & 7;
    int s = src[e], d = dst[e];
    float alpha = ebuf[e] / denom[d];
    float4 fv = ((const float4*)(F + (size_t)s * HF))[j];
    float* op = acc + (size_t)d * HF + j * 4;
    atomicAdd(op + 0, alpha * fv.x);
    atomicAdd(op + 1, alpha * fv.y);
    atomicAdd(op + 2, alpha * fv.z);
    atomicAdd(op + 3, alpha * fv.w);
}

// ------------------------- bias (+optional ReLU), in place ------------------

__global__ void gat_bias_act(float* __restrict__ v, const float* __restrict__ b,
                             int do_relu, int total) {
    int i = blockIdx.x * blockDim.x + threadIdx.x;
    if (i >= total) return;
    float x = v[i] + b[i & (HF - 1)];
    if (do_relu) x = fmaxf(x, 0.f);
    v[i] = x;
}

// ------------------------- driver ------------------------------------------

static void run_layer(const float* X, int KDIM,
                      const float* W, const float* al, const float* ar,
                      const float* b, const int* src, const int* dst,
                      float* feat, float* el, float* er,
                      unsigned* mkey, float* mflt, float* denom, float* ebuf,
                      float* acc_out, int do_relu, hipStream_t stream) {
    const int ntiles = N_NODES / 16;  // 6250, exact
    hipMemsetAsync(acc_out, 0, (size_t)NF * sizeof(float), stream);
    hipMemsetAsync(mkey,    0, (size_t)N_NODES * sizeof(unsigned), stream);
    hipMemsetAsync(denom,   0, (size_t)N_NODES * sizeof(float), stream);

    if (KDIM == 128)
        gat_gemm_wmma<128><<<(ntiles + 7) / 8, 256, 0, stream>>>(X, W, feat, ntiles);
    else
        gat_gemm_wmma<32><<<(ntiles + 7) / 8, 256, 0, stream>>>(X, W, feat, ntiles);

    gat_node_dot<<<(N_NODES + 255) / 256, 256, 0, stream>>>(feat, al, ar, el, er, N_NODES);
    gat_edge_score<<<(N_EDGES + 255) / 256, 256, 0, stream>>>(src, dst, el, er, ebuf, mkey, N_EDGES);
    gat_decode_m<<<(N_NODES + 255) / 256, 256, 0, stream>>>(mkey, mflt, N_NODES);
    gat_edge_expsum<<<(N_EDGES + 255) / 256, 256, 0, stream>>>(dst, ebuf, mflt, denom, N_EDGES);
    gat_edge_scatter<<<(N_EDGES * 8 + 255) / 256, 256, 0, stream>>>(src, dst, ebuf, denom, feat,
                                                                    acc_out, N_EDGES);
    gat_bias_act<<<(NF + 255) / 256, 256, 0, stream>>>(acc_out, b, do_relu, NF);
}

extern "C" void kernel_launch(void* const* d_in, const int* in_sizes, int n_in,
                              void* d_out, int out_size, void* d_ws, size_t ws_size,
                              hipStream_t stream) {
    const float* x   = (const float*)d_in[0];
    const int*   src = (const int*)  d_in[1];
    const int*   dst = (const int*)  d_in[2];
    const float* W1  = (const float*)d_in[3];
    const float* al1 = (const float*)d_in[4];
    const float* ar1 = (const float*)d_in[5];
    const float* b1  = (const float*)d_in[6];
    const float* W2  = (const float*)d_in[7];
    const float* al2 = (const float*)d_in[8];
    const float* ar2 = (const float*)d_in[9];
    const float* b2  = (const float*)d_in[10];
    float* out = (float*)d_out;

    // Workspace layout (floats): feat | h1 | el | er | mkey | mflt | denom | ebuf
    float*    ws    = (float*)d_ws;
    float*    feat  = ws;                       // NF
    float*    h1    = feat + NF;                // NF
    float*    el    = h1 + NF;                  // N
    float*    er    = el + N_NODES;             // N
    unsigned* mkey  = (unsigned*)(er + N_NODES);// N
    float*    mflt  = (float*)(mkey + N_NODES); // N
    float*    denom = mflt + N_NODES;           // N
    float*    ebuf  = denom + N_NODES;          // E

    // Layer 1: x[100000,128] -> h1[100000,32], ReLU applied.
    run_layer(x, 128, W1, al1, ar1, b1, src, dst,
              feat, el, er, mkey, mflt, denom, ebuf, h1, /*relu=*/1, stream);

    // Layer 2: h1 -> d_out, bias only.
    run_layer(h1, 32, W2, al2, ar2, b2, src, dst,
              feat, el, er, mkey, mflt, denom, ebuf, out, /*relu=*/0, stream);
}
